// RegLSTM_27324581937435
// MI455X (gfx1250) — compile-verified, matrix operand order checked by reading, not run on hardware
//
#include <hip/hip_runtime.h>
#include <hip/hip_bf16.h>

typedef __attribute__((ext_vector_type(16))) _Float16 v16h;
typedef __attribute__((ext_vector_type(8)))  _Float16 v8h;
typedef __attribute__((ext_vector_type(8)))  float    v8f;

#define B_   64
#define S_   1024
#define D_   512
#define H_   512
#define G4H  2048
#define NWG_LSTM 32   // 32 WGs * 64 packed cols = 2048 gate columns

static __device__ __forceinline__ v16h frag_cat(v8h lo, v8h hi) {
  return __builtin_shufflevector(lo, hi, 0,1,2,3,4,5,6,7,8,9,10,11,12,13,14,15);
}
static __device__ __forceinline__ v8f wmma_f16(v16h a, v16h b, v8f c) {
  // D = A(16x32 f16) * B(32x16 f16) + C(16x16 f32)
  return __builtin_amdgcn_wmma_f32_16x16x32_f16(false, a, false, b, (short)0, c,
                                                false, false);
}
static __device__ __forceinline__ float sigmoidf_(float x) {
  return 1.0f / (1.0f + __expf(-x));
}

// Per-lane 16B global->LDS async copy (CDNA5 GLOBAL_LOAD_ASYNC_TO_LDS_B128,
// tracked with ASYNCcnt). LDS offset = low 32 bits of the generic address.
static __device__ __forceinline__ void async_g2l_b128(const _Float16* g, void* l) {
  asm volatile("global_load_async_to_lds_b128 %0, %1, off"
               :: "v"((unsigned)(size_t)l), "v"(g)
               : "memory");
}
static __device__ __forceinline__ void wait_async0() {
#if __has_builtin(__builtin_amdgcn_s_wait_asynccnt)
  __builtin_amdgcn_s_wait_asynccnt(0);
#else
  asm volatile("s_wait_asynccnt 0" ::: "memory");
#endif
}

// ---------------------------------------------------------------------------
// Pack W/U into f16, column-major ("fragment friendly": 16B loads give the
// 8-element K-chunks of the B-fragment), with gate-permuted columns:
//   packed col m = g*64 + q*16 + c  <->  original col q*512 + g*16 + c
// so workgroup g of the recurrent kernel owns H-slice [g*16, g*16+16) for all
// four gates. Bias is permuted the same way.
// ---------------------------------------------------------------------------
__global__ void pack_w_kernel(const float* __restrict__ W,
                              const float* __restrict__ U,
                              const float* __restrict__ b,
                              _Float16* __restrict__ Wt,
                              _Float16* __restrict__ Ut,
                              float* __restrict__ bp)
{
  const int k = blockIdx.x * 64 + threadIdx.x;   // 0..511
  const int m = blockIdx.y * 4  + threadIdx.y;   // 0..2047 (packed col)
  const int l = blockIdx.z;                      // layer
  const int g = m >> 6, q = (m >> 4) & 3, c = m & 15;
  const int col = q * H_ + g * 16 + c;           // original gate column
  Wt[((size_t)l * G4H + m) * D_ + k] = (_Float16)W[((size_t)l * D_ + k) * G4H + col];
  Ut[((size_t)l * G4H + m) * H_ + k] = (_Float16)U[((size_t)l * H_ + k) * G4H + col];
  if (k == 0) bp[(size_t)l * G4H + m] = b[(size_t)l * G4H + col];
}

__global__ void cvt_f16_kernel(const float* __restrict__ src,
                               _Float16* __restrict__ dst, int n)
{
  const int i = (blockIdx.x * blockDim.x + threadIdx.x) * 8;
  if (i < n) {
    float4 a = *(const float4*)(src + i);
    float4 c = *(const float4*)(src + i + 4);
    v8h h = { (_Float16)a.x, (_Float16)a.y, (_Float16)a.z, (_Float16)a.w,
              (_Float16)c.x, (_Float16)c.y, (_Float16)c.z, (_Float16)c.w };
    *(v8h*)(dst + i) = h;
  }
}

__global__ void init_kernel(_Float16* __restrict__ h16, unsigned* __restrict__ bar)
{
  const int i = blockIdx.x * blockDim.x + threadIdx.x;
  if (i < B_ * H_) h16[i] = (_Float16)0.0f;
  if (i < S_)      bar[i] = 0u;
}

// ---------------------------------------------------------------------------
// XW = X(f16) * Wt(f16 packed) + bias. M is re-ordered as (s*64 + b) so every
// 16x16 output tile is exactly one LSTM-step fragment; results are stored as
// raw 32-lane x 8 f32 fragments:
//   XWf[(((s*4 + mtile)*128 + ntile)*256) + lane*8 + r]
// WG tile 128M x 64N, 8 waves (4Mx2N), 2x2 WMMA accumulators per wave.
// A tiles are staged via double-buffered GLOBAL_LOAD_ASYNC_TO_LDS_B128 (the
// gather also performs the b-major M reordering); B-fragments load straight
// from the packed L2-resident weights.
// ---------------------------------------------------------------------------
__global__ __launch_bounds__(256)
void gemm_xw_kernel(const _Float16* __restrict__ X,   // [B*S][512] f16, row = b*S+s
                    const _Float16* __restrict__ Wt,  // [2048][512] packed
                    const float* __restrict__ bp,     // [2048] packed
                    float* __restrict__ XWf)          // fragment blob (f32)
{
  __shared__ __align__(16) _Float16 As[2][128][40];   // 32 K + 8 pad per row
  const int tid  = threadIdx.x;
  const int lane = tid & 31, wave = tid >> 5;
  const int m0 = blockIdx.x * 128;                    // transposed-M base
  const int n0 = blockIdx.y * 64;
  const int mw = (wave & 3) * 32;
  const int nw = (wave >> 2) * 32;
  const int l15 = lane & 15;
  const int kc  = (lane >> 4) * 8;                    // K-chunk select
  v8f acc[2][2] = {};

  // staging assignment: 512 16B chunks per tile, 2 per thread
  int srow[2], sj[2];
  const _Float16* sbase[2];
  #pragma unroll
  for (int t = 0; t < 2; ++t) {
    const int cid = tid + t * 256;
    srow[t] = cid >> 2;                               // 0..127 local row
    sj[t]   = cid & 3;                                // 16B chunk within row
    const int rg = m0 + srow[t];                      // global row (s*64 + b)
    const int bb = rg & 63, ss = rg >> 6;
    sbase[t] = X + ((size_t)bb * S_ + ss) * D_ + sj[t] * 8;
  }

  // prologue: stage tile 0 into buffer 0
  #pragma unroll
  for (int t = 0; t < 2; ++t)
    async_g2l_b128(sbase[t], &As[0][srow[t]][sj[t] * 8]);

  int pb = 0;
  for (int ks = 0; ks < D_; ks += 32, pb ^= 1) {
    wait_async0();            // own copies for current tile complete
    __syncthreads();          // everyone's copies done; prior reads of nxt done
    if (ks + 32 < D_) {       // overlap: stage next tile while computing
      #pragma unroll
      for (int t = 0; t < 2; ++t)
        async_g2l_b128(sbase[t] + (ks + 32), &As[pb ^ 1][srow[t]][sj[t] * 8]);
    }
    v16h a[2], bf[2];
    #pragma unroll
    for (int i = 0; i < 2; ++i) {
      const _Float16* ap = &As[pb][mw + i * 16 + l15][kc];
      a[i] = frag_cat(*(const v8h*)ap, *(const v8h*)(ap + 16));
    }
    #pragma unroll
    for (int j = 0; j < 2; ++j) {
      const _Float16* bq = Wt + (size_t)(n0 + nw + j * 16 + l15) * D_ + ks + kc;
      bf[j] = frag_cat(*(const v8h*)bq, *(const v8h*)(bq + 16));
    }
    #pragma unroll
    for (int i = 0; i < 2; ++i)
      #pragma unroll
      for (int j = 0; j < 2; ++j)
        acc[i][j] = wmma_f16(a[i], bf[j], acc[i][j]);
  }

  // bias + fragment-blob store (2x b128 per accumulator)
  #pragma unroll
  for (int i = 0; i < 2; ++i) {
    const int mrow = m0 + mw + i * 16;
    const int ss = mrow >> 6;            // step
    const int mb = (mrow >> 4) & 3;      // batch tile 0..3
    #pragma unroll
    for (int j = 0; j < 2; ++j) {
      const int nt = (n0 + nw) / 16 + j; // gate-column tile 0..127
      const float bias = bp[nt * 16 + l15];
      v8f v = acc[i][j];
      #pragma unroll
      for (int r = 0; r < 8; ++r) v[r] += bias;
      *(v8f*)(XWf + (((size_t)ss * 4 + mb) * 128 + nt) * 256 + lane * 8) = v;
    }
  }
}

// ---------------------------------------------------------------------------
// Persistent recurrent kernel: 32 WGs, each owns 64 packed gate columns
// (= 16 h columns x 4 gates). Per step: gates = h@U (WMMA, K split across
// wave pairs, partials merged via LDS) + XW fragments, then register-local
// i/f/g/o fusion, c/h update, h written back (f16 for next step / next layer
// GEMM + optional f32 layer output), then a device-scope barrier over the 32
// WGs (per-step counter => no reset needed).
// ---------------------------------------------------------------------------
__global__ __launch_bounds__(256)
void lstm_persist_kernel(const _Float16* __restrict__ Ut,  // packed [2048][512]
                         const float* __restrict__ XWf,    // fragment blob
                         _Float16* __restrict__ h16,       // [64][512], zeroed
                         float* __restrict__ Y,            // [B][S][H] f32 or null
                         _Float16* __restrict__ Y16,       // [B][S][H] f16 or null
                         float* __restrict__ outh,         // [64][512] or null
                         float* __restrict__ outc,         // [64][512] or null
                         unsigned* __restrict__ bar)       // [S], zeroed
{
  __shared__ float part[4][4][32][8];            // [Mtile][gate][lane][vgpr]
  const int g    = blockIdx.x;                   // 0..31 -> h cols [g*16, g*16+16)
  const int tid  = threadIdx.x;
  const int lane = tid & 31, wave = tid >> 5;
  const int mt   = wave & 3;                     // batch tile (16 rows)
  const int kh   = wave >> 2;                    // K half: 0 or 1
  const int l15  = lane & 15;
  const int kc   = (lane >> 4) * 8;
  const int rlo  = (lane >> 4) * 8;
  const int hcol = g * 16 + l15;

  v8f c_state = {};                              // c for (mt, lane, r); kh==0 only

  const _Float16* ub[4];
  #pragma unroll
  for (int q = 0; q < 4; ++q)
    ub[q] = Ut + (size_t)(g * 64 + q * 16 + l15) * H_ + kh * 256 + kc;
  const _Float16* arow = h16 + (size_t)(mt * 16 + l15) * H_ + kh * 256 + kc;

  for (int s = 0; s < S_; ++s) {
    v8f acc[4] = {{}, {}, {}, {}};
    #pragma unroll
    for (int q = 0; q < 4; ++q) {
      const _Float16* ap = arow;
      const _Float16* up = ub[q];
      #pragma unroll
      for (int ks = 0; ks < 256; ks += 32) {
        v16h a = frag_cat(*(const v8h*)ap, *(const v8h*)(ap + 16));
        v16h b = frag_cat(*(const v8h*)up, *(const v8h*)(up + 16));
        acc[q] = wmma_f16(a, b, acc[q]);
        ap += 32; up += 32;
      }
    }
    if (kh == 1) {
      #pragma unroll
      for (int q = 0; q < 4; ++q)
        *(v8f*)&part[mt][q][lane][0] = acc[q];
    }
    __syncthreads();
    if (kh == 0) {
      const size_t fbase = (((size_t)s * 4 + mt) * 128 + g * 4) * 256 + lane * 8;
      if (s + 1 < S_)  // warm next step's projection fragments into cache
        __builtin_prefetch(XWf + fbase + (size_t)4 * 128 * 256, 0, 0);
      float gates[4][8];
      #pragma unroll
      for (int q = 0; q < 4; ++q) {
        v8f other = *(const v8f*)&part[mt][q][lane][0];
        v8f xwv = *(const v8f*)(XWf + fbase + (size_t)q * 256);
        #pragma unroll
        for (int r = 0; r < 8; ++r)
          gates[q][r] = acc[q][r] + other[r] + xwv[r];
      }
      #pragma unroll
      for (int r = 0; r < 8; ++r) {
        const int brow = mt * 16 + rlo + r;
        const float iv = sigmoidf_(gates[0][r]);
        const float fv = sigmoidf_(gates[1][r]);
        const float gv = tanhf(gates[2][r]);
        const float ov = sigmoidf_(gates[3][r]);
        const float cv = fv * c_state[r] + iv * gv;
        const float hv = ov * tanhf(cv);
        c_state[r] = cv;
        h16[(size_t)brow * H_ + hcol] = (_Float16)hv;
        if (Y)   Y[((size_t)brow * S_ + s) * H_ + hcol] = hv;
        if (Y16) Y16[((size_t)brow * S_ + s) * H_ + hcol] = (_Float16)hv;
        if (outh && s == S_ - 1) {
          outh[(size_t)brow * H_ + hcol] = hv;
          outc[(size_t)brow * H_ + hcol] = cv;
        }
      }
    }
    // device-scope barrier across the 32 persistent workgroups
    __syncthreads();
    if (tid == 0) {
      __hip_atomic_fetch_add(&bar[s], 1u, __ATOMIC_RELEASE, __HIP_MEMORY_SCOPE_AGENT);
      while (__hip_atomic_load(&bar[s], __ATOMIC_ACQUIRE, __HIP_MEMORY_SCOPE_AGENT)
             < (unsigned)NWG_LSTM)
        __builtin_amdgcn_s_sleep(1);
    }
    __syncthreads();
  }
}

// ---------------------------------------------------------------------------
extern "C" void kernel_launch(void* const* d_in, const int* in_sizes, int n_in,
                              void* d_out, int out_size, void* d_ws, size_t ws_size,
                              hipStream_t stream)
{
  (void)in_sizes; (void)n_in; (void)out_size; (void)ws_size;
  const float* x = (const float*)d_in[0];
  const float* W = (const float*)d_in[1];
  const float* U = (const float*)d_in[2];
  const float* b = (const float*)d_in[3];

  float* out   = (float*)d_out;
  float* out_y = out;                               // [B,S,H]
  float* out_h = out + (size_t)B_ * S_ * H_;        // [B,H]
  float* out_c = out_h + (size_t)B_ * H_;           // [B,H]

  char* ws = (char*)d_ws;
  _Float16* Wt  = (_Float16*)ws; ws += (size_t)2 * G4H * D_ * sizeof(_Float16);
  _Float16* Ut  = (_Float16*)ws; ws += (size_t)2 * G4H * H_ * sizeof(_Float16);
  float*    bp  = (float*)ws;    ws += (size_t)2 * G4H * sizeof(float);
  float*    xwf = (float*)ws;    ws += (size_t)B_ * S_ * G4H * sizeof(float);
  _Float16* x16 = (_Float16*)ws; ws += (size_t)B_ * S_ * D_ * sizeof(_Float16);
  _Float16* y16 = (_Float16*)ws; ws += (size_t)B_ * S_ * H_ * sizeof(_Float16);
  _Float16* h16 = (_Float16*)ws; ws += (size_t)B_ * H_ * sizeof(_Float16);
  unsigned* bar = (unsigned*)ws; ws += (size_t)S_ * sizeof(unsigned);

  pack_w_kernel<<<dim3(D_ / 64, G4H / 4, 2), dim3(64, 4, 1), 0, stream>>>(
      W, U, b, Wt, Ut, bp);
  cvt_f16_kernel<<<(B_ * S_ * D_) / (256 * 8), 256, 0, stream>>>(
      x, x16, B_ * S_ * D_);

  for (int layer = 0; layer < 2; ++layer) {
    const _Float16* xin = (layer == 0) ? x16 : y16;
    gemm_xw_kernel<<<dim3((B_ * S_) / 128, G4H / 64), 256, 0, stream>>>(
        xin, Wt + (size_t)layer * G4H * D_, bp + (size_t)layer * G4H, xwf);
    init_kernel<<<(B_ * H_ + 255) / 256, 256, 0, stream>>>(h16, bar);
    lstm_persist_kernel<<<NWG_LSTM, 256, 0, stream>>>(
        Ut + (size_t)layer * G4H * H_, xwf, h16,
        (layer == 1) ? out_y : nullptr,          // f32 sequence output
        (layer == 0) ? y16 : nullptr,            // f16 feed to next layer's GEMM
        (layer == 1) ? out_h : nullptr, (layer == 1) ? out_c : nullptr, bar);
  }
}
